// GCN_58067957842339
// MI455X (gfx1250) — matrix operand order repaired and checked by reading, compiled, not run on hardware
//
#include <hip/hip_runtime.h>
#include <hip/hip_bf16.h>

typedef __attribute__((ext_vector_type(16))) __bf16 v16bf;
typedef __attribute__((ext_vector_type(8)))  __bf16 v8bf;
typedef __attribute__((ext_vector_type(4)))  __bf16 v4bf;
typedef __attribute__((ext_vector_type(8)))  float  v8f;

union V16U { v16bf v; v8bf h[2]; };

#define K_DIM 128

// ---------------------------------------------------------------- degree/norm
__global__ __launch_bounds__(256) void k_init_deg(float* deg, int n) {
    int i = blockIdx.x * 256 + threadIdx.x;
    if (i < n) deg[i] = 1.0f;  // self-loop
}

__global__ __launch_bounds__(256) void k_deg_accum(const int* __restrict__ dst,
                                                   float* deg, int e) {
    int i = blockIdx.x * 256 + threadIdx.x;
    if (i < e) atomicAdd(&deg[dst[i]], 1.0f);
}

__global__ __launch_bounds__(256) void k_norm(const float* __restrict__ deg,
                                              float* __restrict__ nrm, int n) {
    int i = blockIdx.x * 256 + threadIdx.x;
    if (i < n) nrm[i] = rsqrtf(deg[i]);
}

// ---------------------------------------------------------------- bf16 WMMA GEMM
// H[n, NCOL] = X[n, 128] @ W[128, NCOL]   (no bias; f32 accumulate)
// 256 threads = 8 waves; block covers 128 rows; wave w covers rows w*16..w*16+15.
// A fragment = two contiguous 16B runs of a row; B fragment = two contiguous
// 16B runs of a column of W (staged transposed in LDS). Padded LDS rows avoid
// 16-way bank conflicts on the per-lane row-strided fragment loads.
template <int NCOL>
__global__ __launch_bounds__(256) void k_gemm_bf16(const float* __restrict__ X,
                                                   const float* __restrict__ W,
                                                   float* __restrict__ H, int n) {
    constexpr int WROW = K_DIM + 8;   // padded row stride of WbT (bf16 units)
    constexpr int AROW = 32 + 8;      // padded row stride of As  (bf16 units)
    __shared__ __bf16 WbT[NCOL * WROW];   // W transposed: WbT[c][k]
    __shared__ __bf16 As[128 * AROW];     // X chunk: 128 rows x 32 K

    const int tid  = threadIdx.x;
    const int wave = tid >> 5;
    const int lane = tid & 31;
    const int m    = lane & 15;   // row (A) / col (B) within 16x16 tile
    const int half = lane >> 4;   // lane half selects K sub-range
    const int rowBase = blockIdx.x * 128;
    const bool fullBlock = (rowBase + 128) <= n;

    // Stage W transposed into LDS as bf16: float4 global reads (coalesced),
    // scattered 2B LDS stores (one-time transpose).
    constexpr int WQ = K_DIM * NCOL / 4;  // number of float4 elements
    for (int q = tid; q < WQ; q += 256) {
        int k = q / (NCOL / 4);
        int c = (q % (NCOL / 4)) * 4;
        float4 f = *(const float4*)(W + (size_t)k * NCOL + c);
        WbT[(c + 0) * WROW + k] = (__bf16)f.x;
        WbT[(c + 1) * WROW + k] = (__bf16)f.y;
        WbT[(c + 2) * WROW + k] = (__bf16)f.z;
        WbT[(c + 3) * WROW + k] = (__bf16)f.w;
    }

    v8f acc[NCOL / 16];
    v8f zero = {};
#pragma unroll
    for (int t = 0; t < NCOL / 16; ++t) acc[t] = zero;

    const int mLoc = wave * 16 + m;

    for (int kc = 0; kc < K_DIM; kc += 32) {
        __syncthreads();
        // Stage 128x32 X chunk: 4096 floats = 256 threads x 4 float4.
        // Batch loads into registers first so VMEM issues overlap, then
        // convert + store to LDS.
        float4 f[4];
        int rr[4], cc[4];
#pragma unroll
        for (int u = 0; u < 4; ++u) {
            int i4 = tid + u * 256;
            rr[u] = i4 >> 3;
            cc[u] = (i4 & 7) * 4;
        }
        if (fullBlock) {
#pragma unroll
            for (int u = 0; u < 4; ++u)
                f[u] = *(const float4*)(X + (size_t)(rowBase + rr[u]) * K_DIM + kc + cc[u]);
        } else {
#pragma unroll
            for (int u = 0; u < 4; ++u) {
                int gr = rowBase + rr[u];
                f[u] = make_float4(0.f, 0.f, 0.f, 0.f);
                if (gr < n)
                    f[u] = *(const float4*)(X + (size_t)gr * K_DIM + kc + cc[u]);
            }
        }
#pragma unroll
        for (int u = 0; u < 4; ++u) {
            v4bf o;
            o[0] = (__bf16)f[u].x; o[1] = (__bf16)f[u].y;
            o[2] = (__bf16)f[u].z; o[3] = (__bf16)f[u].w;
            *(v4bf*)(As + rr[u] * AROW + cc[u]) = o;
        }
        __syncthreads();

        // A fragment: elements 0..7  = As[mLoc][half*8 .. half*8+7]
        //             elements 8..15 = As[mLoc][16+half*8 .. 16+half*8+7]
        V16U a;
        const __bf16* ap = As + mLoc * AROW + half * 8;
        a.h[0] = *(const v8bf*)(ap);
        a.h[1] = *(const v8bf*)(ap + 16);

#pragma unroll
        for (int t = 0; t < NCOL / 16; ++t) {
            // B fragment: elements 0..7 = WbT[t*16+m][kc + half*8 .. +7], 8..15 = WbT[t*16+m][kc + 16 + half*8 .. +7]
            V16U b;
            const __bf16* bp = WbT + (t * 16 + m) * WROW + kc + half * 8;
            b.h[0] = *(const v8bf*)(bp);
            b.h[1] = *(const v8bf*)(bp + 16);
            acc[t] = __builtin_amdgcn_wmma_f32_16x16x32_bf16(
                false, a.v, false, b.v, (short)0, acc[t], false, false);
        }
    }

    // C/D layout: VGPR r holds row (r + 8*half), col = m.
#pragma unroll
    for (int t = 0; t < NCOL / 16; ++t) {
#pragma unroll
        for (int r = 0; r < 8; ++r) {
            int row = rowBase + wave * 16 + r + 8 * half;
            if (row < n) H[(size_t)row * NCOL + t * 16 + m] = acc[t][r];
        }
    }
}

// ---------------------------------------------------------------- self-loop init
// agg[i] = H[i] * norm[node]^2   (flat over n*F)
template <int F>
__global__ __launch_bounds__(256) void k_selfloop(const float* __restrict__ H,
                                                  const float* __restrict__ nrm,
                                                  float* __restrict__ agg, int n) {
    size_t i = (size_t)blockIdx.x * 256 + threadIdx.x;
    size_t total = (size_t)n * F;
    if (i < total) {
        float nv = nrm[i / F];
        agg[i] = H[i] * nv * nv;
    }
}

// ---------------------------------------------------------------- edge scatter
// One wave32 per edge. F=128: float4/lane; F=64: float2/lane.
// agg destination rows live in the 192MB L2, so atomics resolve on-chip.
template <int F>
__global__ __launch_bounds__(256) void k_aggregate(const float* __restrict__ H,
                                                   const int* __restrict__ src,
                                                   const int* __restrict__ dst,
                                                   const float* __restrict__ nrm,
                                                   float* __restrict__ agg, int e) {
    int wId  = (blockIdx.x * 256 + threadIdx.x) >> 5;
    int lane = threadIdx.x & 31;
    if (wId >= e) return;
    int s = src[wId], d = dst[wId];
    float coef = nrm[s] * nrm[d];
    if (F == 128) {
        const float4* hp = (const float4*)(H + (size_t)s * F);
        float4 v = hp[lane];
        float* ap = agg + (size_t)d * F + lane * 4;
        atomicAdd(ap + 0, v.x * coef);
        atomicAdd(ap + 1, v.y * coef);
        atomicAdd(ap + 2, v.z * coef);
        atomicAdd(ap + 3, v.w * coef);
    } else {
        const float2* hp = (const float2*)(H + (size_t)s * F);
        float2 v = hp[lane];
        float* ap = agg + (size_t)d * F + lane * 2;
        atomicAdd(ap + 0, v.x * coef);
        atomicAdd(ap + 1, v.y * coef);
    }
}

// ---------------------------------------------------------------- bias + relu
template <int F>
__global__ __launch_bounds__(256) void k_bias_relu(float* __restrict__ z,
                                                   const float* __restrict__ b, int n) {
    size_t i = (size_t)blockIdx.x * 256 + threadIdx.x;
    if (i < (size_t)n * F) {
        float v = z[i] + b[i % F];
        z[i] = v > 0.0f ? v : 0.0f;
    }
}

// ---------------------------------------------------------------- softmax(64)
// One wave32 per node, 2 values per lane.
__global__ __launch_bounds__(256) void k_softmax64(const float* __restrict__ agg,
                                                   const float* __restrict__ b,
                                                   float* __restrict__ out, int n) {
    int node = (blockIdx.x * 256 + threadIdx.x) >> 5;
    int lane = threadIdx.x & 31;
    if (node >= n) return;
    float2 v = *(const float2*)(agg + (size_t)node * 64 + lane * 2);
    v.x += b[lane * 2];
    v.y += b[lane * 2 + 1];
    float mx = fmaxf(v.x, v.y);
#pragma unroll
    for (int off = 16; off > 0; off >>= 1) mx = fmaxf(mx, __shfl_xor(mx, off, 32));
    float e0 = __expf(v.x - mx), e1 = __expf(v.y - mx);
    float s = e0 + e1;
#pragma unroll
    for (int off = 16; off > 0; off >>= 1) s += __shfl_xor(s, off, 32);
    float inv = 1.0f / s;
    float2 o;
    o.x = e0 * inv;
    o.y = e1 * inv;
    *(float2*)(out + (size_t)node * 64 + lane * 2) = o;
}

// ---------------------------------------------------------------- launcher
extern "C" void kernel_launch(void* const* d_in, const int* in_sizes, int n_in,
                              void* d_out, int out_size, void* d_ws, size_t ws_size,
                              hipStream_t stream) {
    const float* x  = (const float*)d_in[0];
    const int*   ei = (const int*)d_in[1];
    const float* W1 = (const float*)d_in[2];
    const float* b1 = (const float*)d_in[3];
    const float* W2 = (const float*)d_in[4];
    const float* b2 = (const float*)d_in[5];

    const int N = in_sizes[0] / 128;
    const int E = in_sizes[1] / 2;
    const int* src = ei;
    const int* dst = ei + E;

    // workspace layout (256B aligned slots); h2/agg2 reuse h1's region
    size_t nPad = ((size_t)N + 63) & ~(size_t)63;
    float* deg  = (float*)d_ws;
    float* nrm  = deg + nPad;
    float* h1   = nrm + nPad;
    float* agg1 = h1 + (size_t)N * 128;  // becomes z1 after bias+relu
    float* h2   = h1;                    // reuse (h1 dead after layer-1 aggregate)
    float* agg2 = h1 + (size_t)N * 64;   // second half of old h1 region
    float* out  = (float*)d_out;

    auto grid1 = [](size_t c) { return (unsigned)((c + 255) / 256); };

    // degree + norm
    k_init_deg<<<grid1(N), 256, 0, stream>>>(deg, N);
    k_deg_accum<<<grid1(E), 256, 0, stream>>>(dst, deg, E);
    k_norm<<<grid1(N), 256, 0, stream>>>(deg, nrm, N);

    // layer 1: h1 = x @ W1 ; agg1 = self + edges ; z1 = relu(agg1 + b1)
    k_gemm_bf16<128><<<(N + 127) / 128, 256, 0, stream>>>(x, W1, h1, N);
    k_selfloop<128><<<grid1((size_t)N * 128), 256, 0, stream>>>(h1, nrm, agg1, N);
    k_aggregate<128><<<(E + 7) / 8, 256, 0, stream>>>(h1, src, dst, nrm, agg1, E);
    k_bias_relu<128><<<grid1((size_t)N * 128), 256, 0, stream>>>(agg1, b1, N);

    // layer 2: h2 = z1 @ W2 ; agg2 = self + edges ; out = softmax(agg2 + b2)
    k_gemm_bf16<64><<<(N + 127) / 128, 256, 0, stream>>>(agg1, W2, h2, N);
    k_selfloop<64><<<grid1((size_t)N * 64), 256, 0, stream>>>(h2, nrm, agg2, N);
    k_aggregate<64><<<(E + 7) / 8, 256, 0, stream>>>(h2, src, dst, nrm, agg2, E);
    k_softmax64<<<(N + 7) / 8, 256, 0, stream>>>(agg2, b2, out, N);
}
